// MixtralSparseMoeBlock_75960791597570
// MI455X (gfx1250) — compile-verified, matrix operand order checked by reading, not run on hardware
//
#include <hip/hip_runtime.h>
#include <hip/hip_fp16.h>
#include <stdint.h>

typedef __attribute__((ext_vector_type(16))) _Float16 v16h;
typedef __attribute__((ext_vector_type(8)))  _Float16 v8h;
typedef __attribute__((ext_vector_type(8)))  float    v8f;

#define H_DIM 1024
#define F_DIM 3584
#define E_NUM 8
#define BK    64          // K-step per LDS tile
#define LDT   72          // LDS row stride in halves (144B: 16B-aligned, bank-spread)

// Load one 16x32 f16 WMMA operand fragment from an LDS tile row block.
// Per ISA 7.12.2: lanes 0-15 hold K=0..7 (VGPR0-3) and K=16..23 (VGPR4-7);
// lanes 16-31 hold K=8..15 and K=24..31. Row = lane%16.
__device__ __forceinline__ v16h ldfrag(const _Float16* t, int lane) {
    int r  = lane & 15;
    int kb = (lane >> 4) * 8;
    const _Float16* p = t + r * LDT + kb;
    union { v16h v; v8h h[2]; } f;
    f.h[0] = *(const v8h*)(p);
    f.h[1] = *(const v8h*)(p + 16);
    return f.v;
}

__device__ __forceinline__ v8h cvt8(float4 a, float4 b) {
    return (v8h){(_Float16)a.x, (_Float16)a.y, (_Float16)a.z, (_Float16)a.w,
                 (_Float16)b.x, (_Float16)b.y, (_Float16)b.z, (_Float16)b.w};
}

__device__ __forceinline__ v8f wmma16(v16h a, v16h b, v8f c) {
    return __builtin_amdgcn_wmma_f32_16x16x32_f16(false, a, false, b, (short)0, c, false, false);
}

__global__ __launch_bounds__(256)
void zero_init(float* __restrict__ out, int n, int* __restrict__ list, int nl,
               int* __restrict__ counts) {
    int i = blockIdx.x * blockDim.x + threadIdx.x;
    int stride = gridDim.x * blockDim.x;
    for (int j = i; j < n; j += stride) out[j] = 0.f;
    for (int j = i; j < nl; j += stride) list[j] = 0;
    if (i < E_NUM) counts[i] = 0;
}

// One wave per token: logits, top-2, renormalized softmax weights, f16 copy of x.
__global__ __launch_bounds__(256)
void router_kernel(const float* __restrict__ x, const float* __restrict__ gw,
                   _Float16* __restrict__ xb, int* __restrict__ counts,
                   float* __restrict__ rw, int2* __restrict__ eidx, int T) {
    int wave = threadIdx.x >> 5;
    int lane = threadIdx.x & 31;
    int t = blockIdx.x * 8 + wave;
    if (t >= T) return;
    const float* xr = x + (size_t)t * H_DIM;
    float acc[E_NUM];
#pragma unroll
    for (int e = 0; e < E_NUM; ++e) acc[e] = 0.f;
    for (int h = lane; h < H_DIM; h += 32) {
        float xv = xr[h];
        xb[(size_t)t * H_DIM + h] = (_Float16)xv;
#pragma unroll
        for (int e = 0; e < E_NUM; ++e) acc[e] += xv * gw[e * H_DIM + h];
    }
#pragma unroll
    for (int off = 16; off > 0; off >>= 1) {
#pragma unroll
        for (int e = 0; e < E_NUM; ++e) acc[e] += __shfl_xor(acc[e], off, 32);
    }
    if (lane == 0) {
        int e0 = 0;
#pragma unroll
        for (int e = 1; e < E_NUM; ++e) if (acc[e] > acc[e0]) e0 = e;
        int e1 = (e0 == 0) ? 1 : 0;
#pragma unroll
        for (int e = 0; e < E_NUM; ++e) if (e != e0 && acc[e] > acc[e1]) e1 = e;
        // renormalized top-2 softmax == sigmoid over logit gap
        float w0 = 1.f / (1.f + __expf(acc[e1] - acc[e0]));
        rw[t * E_NUM + e0] = w0;
        rw[t * E_NUM + e1] = 1.f - w0;
        eidx[t] = make_int2(e0, e1);
        atomicAdd(&counts[e0], 1);
        atomicAdd(&counts[e1], 1);
    }
}

__global__ void prefix_kernel(const int* __restrict__ counts, int* __restrict__ offsets,
                              int* __restrict__ cursors) {
    if (threadIdx.x == 0 && blockIdx.x == 0) {
        int s = 0;
        for (int e = 0; e < E_NUM; ++e) { offsets[e] = s; cursors[e] = s; s += counts[e]; }
        offsets[E_NUM] = s;
    }
}

__global__ __launch_bounds__(256)
void assign_kernel(const int2* __restrict__ eidx, int* cursors, int* __restrict__ list, int T) {
    int t = blockIdx.x * 256 + threadIdx.x;
    if (t >= T) return;
    int2 e = eidx[t];
    int s0 = atomicAdd(&cursors[e.x], 1);
    list[s0] = t;
    int s1 = atomicAdd(&cursors[e.y], 1);
    list[s1] = t;
}

// act[slot, f] = silu(x @ w1[e]^T) * (x @ w3[e]^T), gathered per expert.
// Block: 256 thr = 8 waves; macro-tile 64 tokens x 64 F-cols; K-step 64; double-buffered LDS.
// Wave (wm in 0..1, wn in 0..3) computes rows wm*32..+31, cols wn*16..+15, for both w1 and w3.
__global__ __launch_bounds__(256)
void moe_gemm1(const float* __restrict__ w1, const float* __restrict__ w3,
               const _Float16* __restrict__ xb, const int* __restrict__ counts,
               const int* __restrict__ offsets, const int* __restrict__ list,
               _Float16* __restrict__ act, int maxTiles) {
    __shared__ __align__(16) _Float16 As [2][64 * LDT];
    __shared__ __align__(16) _Float16 B1s[2][64 * LDT];
    __shared__ __align__(16) _Float16 B3s[2][64 * LDT];
    __shared__ int toks[64];

    int e  = blockIdx.y / maxTiles;
    int mt = blockIdx.y % maxTiles;
    int cnt = counts[e];
    if (mt * 64 >= cnt) return;
    int base  = offsets[e];
    int fBase = blockIdx.x * 64;
    int tid   = threadIdx.x;

    if (tid < 64) toks[tid] = list[base + mt * 64 + tid];
    __syncthreads();

    // load mapping: 64 rows x 64 elems per tile, 4 threads/row, 16 elems/thread
    int lr = tid >> 2;             // 0..63
    int lc = (tid & 3) * 16;       // 0,16,32,48
    const _Float16* aSrc = xb + (size_t)toks[lr] * H_DIM + lc;
    size_t wOff = ((size_t)e * F_DIM + fBase + lr) * H_DIM + lc;
    const float* b1Src = w1 + wOff;
    const float* b3Src = w3 + wOff;

    int w = tid >> 5, lane = tid & 31;
    int wm = w & 1, wn = w >> 1;
    v8f acc00 = {}, acc01 = {}, acc10 = {}, acc11 = {};

    uint4  ra[2];
    float4 rb1[4], rb3[4];
    auto loadRegs = [&](int k0) {
        ra[0] = *(const uint4*)(aSrc + k0);
        ra[1] = *(const uint4*)(aSrc + k0 + 8);
#pragma unroll
        for (int i = 0; i < 4; ++i) rb1[i] = *(const float4*)(b1Src + k0 + i * 4);
#pragma unroll
        for (int i = 0; i < 4; ++i) rb3[i] = *(const float4*)(b3Src + k0 + i * 4);
    };
    auto storeRegs = [&](int buf) {
        *(uint4*)&As[buf][lr * LDT + lc]     = ra[0];
        *(uint4*)&As[buf][lr * LDT + lc + 8] = ra[1];
        *(v8h*)&B1s[buf][lr * LDT + lc]      = cvt8(rb1[0], rb1[1]);
        *(v8h*)&B1s[buf][lr * LDT + lc + 8]  = cvt8(rb1[2], rb1[3]);
        *(v8h*)&B3s[buf][lr * LDT + lc]      = cvt8(rb3[0], rb3[1]);
        *(v8h*)&B3s[buf][lr * LDT + lc + 8]  = cvt8(rb3[2], rb3[3]);
    };

    const int NK = H_DIM / BK;
    loadRegs(0);
    storeRegs(0);
    __syncthreads();

    for (int it = 0; it < NK; ++it) {
        int cur = it & 1, nxt = cur ^ 1;
        if (it + 1 < NK) {
            loadRegs((it + 1) * BK);
            if (it + 2 < NK) {          // pull next-next weight tiles toward L2
                __builtin_prefetch(b1Src + (it + 2) * BK, 0, 3);
                __builtin_prefetch(b3Src + (it + 2) * BK, 0, 3);
            }
        }
#pragma unroll
        for (int ks = 0; ks < 2; ++ks) {
            v16h a0 = ldfrag(&As [cur][(wm * 32)      * LDT + ks * 32], lane);
            v16h a1 = ldfrag(&As [cur][(wm * 32 + 16) * LDT + ks * 32], lane);
            v16h b1 = ldfrag(&B1s[cur][(wn * 16)      * LDT + ks * 32], lane);
            v16h b3 = ldfrag(&B3s[cur][(wn * 16)      * LDT + ks * 32], lane);
            acc00 = wmma16(a0, b1, acc00);
            acc01 = wmma16(a0, b3, acc01);
            acc10 = wmma16(a1, b1, acc10);
            acc11 = wmma16(a1, b3, acc11);
        }
        if (it + 1 < NK) storeRegs(nxt);
        __syncthreads();
    }

    int hl = lane >> 4, n = lane & 15;
#pragma unroll
    for (int mi = 0; mi < 2; ++mi) {
        v8f& A1 = mi ? acc10 : acc00;
        v8f& A3 = mi ? acc11 : acc01;
#pragma unroll
        for (int r = 0; r < 8; ++r) {
            int ml = wm * 32 + mi * 16 + hl * 8 + r;  // C layout: M = vgpr + 8*(lane/16)
            int mg = mt * 64 + ml;
            if (mg < cnt) {
                float a1v = A1[r], a3v = A3[r];
                float s = a1v / (1.f + __expf(-a1v));
                act[(size_t)(base + mg) * F_DIM + fBase + wn * 16 + n] = (_Float16)(s * a3v);
            }
        }
    }
}

// out[t, h] += rw[t,e] * (act @ w2[e]^T); scatter via atomicAdd onto zeroed output.
__global__ __launch_bounds__(256)
void moe_gemm2(const float* __restrict__ w2, const _Float16* __restrict__ act,
               const int* __restrict__ counts, const int* __restrict__ offsets,
               const int* __restrict__ list, const float* __restrict__ rw,
               float* __restrict__ out, int maxTiles) {
    __shared__ __align__(16) _Float16 As[2][64 * LDT];
    __shared__ __align__(16) _Float16 Bs[2][64 * LDT];
    __shared__ int   toks[64];
    __shared__ float rws[64];

    int e  = blockIdx.y / maxTiles;
    int mt = blockIdx.y % maxTiles;
    int cnt = counts[e];
    if (mt * 64 >= cnt) return;
    int base  = offsets[e];
    int hBase = blockIdx.x * 64;
    int tid   = threadIdx.x;

    if (tid < 64) {
        int tk = list[base + mt * 64 + tid];
        toks[tid] = tk;
        rws[tid]  = rw[tk * E_NUM + e];
    }
    __syncthreads();

    int lr = tid >> 2;
    int lc = (tid & 3) * 16;
    const _Float16* aSrc = act + (size_t)(base + mt * 64 + lr) * F_DIM + lc;
    const float* bSrc = w2 + ((size_t)e * H_DIM + hBase + lr) * F_DIM + lc;

    int w = tid >> 5, lane = tid & 31;
    int wm = w & 1, wn = w >> 1;
    v8f acc0 = {}, acc1 = {};

    uint4  ra[2];
    float4 rb[4];
    auto loadRegs = [&](int k0) {
        ra[0] = *(const uint4*)(aSrc + k0);
        ra[1] = *(const uint4*)(aSrc + k0 + 8);
#pragma unroll
        for (int i = 0; i < 4; ++i) rb[i] = *(const float4*)(bSrc + k0 + i * 4);
    };
    auto storeRegs = [&](int buf) {
        *(uint4*)&As[buf][lr * LDT + lc]     = ra[0];
        *(uint4*)&As[buf][lr * LDT + lc + 8] = ra[1];
        *(v8h*)&Bs[buf][lr * LDT + lc]       = cvt8(rb[0], rb[1]);
        *(v8h*)&Bs[buf][lr * LDT + lc + 8]   = cvt8(rb[2], rb[3]);
    };

    const int NK = F_DIM / BK;
    loadRegs(0);
    storeRegs(0);
    __syncthreads();

    for (int it = 0; it < NK; ++it) {
        int cur = it & 1, nxt = cur ^ 1;
        if (it + 1 < NK) {
            loadRegs((it + 1) * BK);
            if (it + 2 < NK) __builtin_prefetch(bSrc + (it + 2) * BK, 0, 3);
        }
#pragma unroll
        for (int ks = 0; ks < 2; ++ks) {
            v16h a0 = ldfrag(&As[cur][(wm * 32)      * LDT + ks * 32], lane);
            v16h a1 = ldfrag(&As[cur][(wm * 32 + 16) * LDT + ks * 32], lane);
            v16h bf = ldfrag(&Bs[cur][(wn * 16)      * LDT + ks * 32], lane);
            acc0 = wmma16(a0, bf, acc0);
            acc1 = wmma16(a1, bf, acc1);
        }
        if (it + 1 < NK) storeRegs(nxt);
        __syncthreads();
    }

    int hl = lane >> 4, n = lane & 15;
#pragma unroll
    for (int mi = 0; mi < 2; ++mi) {
        v8f& A = mi ? acc1 : acc0;
#pragma unroll
        for (int r = 0; r < 8; ++r) {
            int ml = wm * 32 + mi * 16 + hl * 8 + r;
            int mg = mt * 64 + ml;
            if (mg < cnt) {
                atomicAdd(&out[(size_t)toks[ml] * H_DIM + hBase + wn * 16 + n],
                          rws[ml] * A[r]);
            }
        }
    }
}

extern "C" void kernel_launch(void* const* d_in, const int* in_sizes, int n_in,
                              void* d_out, int out_size, void* d_ws, size_t ws_size,
                              hipStream_t stream) {
    const float* x  = (const float*)d_in[0];   // [T, H]
    const float* gw = (const float*)d_in[1];   // [E, H]
    const float* w1 = (const float*)d_in[2];   // [E, F, H]
    const float* w3 = (const float*)d_in[3];   // [E, F, H]
    const float* w2 = (const float*)d_in[4];   // [E, H, F]
    float* out = (float*)d_out;

    int T = in_sizes[0] / H_DIM;
    int maxTiles = (T + 63) / 64;
    int nSlots = 2 * T;   // total token-expert assignments (top-2)

    uint8_t* p = (uint8_t*)d_ws;
    auto alloc = [&](size_t bytes) {
        uint8_t* q = p;
        p += (bytes + 255) & ~(size_t)255;
        return q;
    };
    int*      counts  = (int*)alloc(E_NUM * 4);
    int*      offsets = (int*)alloc((E_NUM + 1) * 4);
    int*      cursors = (int*)alloc(E_NUM * 4);
    int*      list    = (int*)alloc((size_t)(nSlots + 64) * 4);
    int2*     eidx    = (int2*)alloc((size_t)T * 8);
    float*    rw      = (float*)alloc((size_t)T * E_NUM * 4);
    _Float16* xb      = (_Float16*)alloc((size_t)T * H_DIM * 2);
    _Float16* act     = (_Float16*)alloc((size_t)(nSlots + 64) * F_DIM * 2);

    zero_init<<<2048, 256, 0, stream>>>(out, T * H_DIM, list, nSlots + 64, counts);
    router_kernel<<<(T + 7) / 8, 256, 0, stream>>>(x, gw, xb, counts, rw, eidx, T);
    prefix_kernel<<<1, 32, 0, stream>>>(counts, offsets, cursors);
    assign_kernel<<<(T + 255) / 256, 256, 0, stream>>>(eidx, cursors, list, T);

    dim3 g1(F_DIM / 64, E_NUM * maxTiles);
    moe_gemm1<<<g1, 256, 0, stream>>>(w1, w3, xb, counts, offsets, list, act, maxTiles);
    dim3 g2(H_DIM / 64, E_NUM * maxTiles);
    moe_gemm2<<<g2, 256, 0, stream>>>(w2, act, counts, offsets, list, rw, out, maxTiles);
}
